// GatedGraphConv_88794153877687
// MI455X (gfx1250) — matrix operand order repaired and checked by reading, not compiled, not run on hardware
//
#include <hip/hip_runtime.h>
#include <hip/hip_bf16.h>

#define N_NODES 100000
#define EDGES   1600000
#define F_IN_   32
#define C_      128
#define L_      5
#define G_      1024

typedef __attribute__((ext_vector_type(16))) __bf16 v16bf;
typedef __attribute__((ext_vector_type(8)))  float  v8f;

__device__ __forceinline__ unsigned short f32_to_bf16_bits(float f) {
    unsigned int u = __float_as_uint(f);
    return (unsigned short)((u + 0x7FFFu + ((u >> 16) & 1u)) >> 16);
}

// ---------------------------------------------------------------------------
// Packed-B layout (global memory, bf16): frag[tile][kstep][lane][16 halves],
// where u[i] = B[kstep*32 + (lane>>4)*8 + (i&7) + (i>>3)*16][tile*16 + (lane&15)]  (ISA 7.12.2
// B-matrix 32x16 layout). Each wave then loads one contiguous 32B per lane.
// ---------------------------------------------------------------------------
__global__ void k_packB(const float* __restrict__ src, unsigned short* __restrict__ dst,
                        int ntiles, int ksteps, int srcld, int transposed, int total) {
    int idx = blockIdx.x * blockDim.x + threadIdx.x;
    if (idx >= total) return;
    int i    = idx & 15;
    int lane = (idx >> 4) & 31;
    int t    = idx >> 9;
    int kstep = t % ksteps;
    int tile  = t / ksteps;
    int k = kstep * 32 + ((lane >> 4) << 3) + (i & 7) + ((i >> 3) << 4);
    int n = tile * 16 + (lane & 15);
    float v = transposed ? src[(size_t)n * srcld + k]    // src [N][K] (e.g. W_ih)
                         : src[(size_t)k * srcld + n];   // src [K][N]
    dst[idx] = f32_to_bf16_bits(v);
}

// ---------------------------------------------------------------------------
// Cooperative A-tile stage: fp32 global -> bf16 A-fragments in LDS.
// A-frag (ISA 7.12.2, 16-bit A 16x32): row = lane&15, koff = (lane>>4)*8,
// u[0..7] = K koff+0..7, u[8..15] = K koff+16..23  -> two contiguous 8-runs.
// One fill unit = (kstep, lane, hi): 8 contiguous fp32 loads -> 8 halves (16B).
// ---------------------------------------------------------------------------
__device__ __forceinline__ void stage_a_unit(const float* __restrict__ Arow /* + row*C */,
                                             unsigned short* __restrict__ sp /* frag base */,
                                             int u /* unit within one tile: [0,256) */) {
    int hi    = u & 1;
    int lane  = (u >> 1) & 31;
    int kstep = (u >> 6) & 3;
    int row   = lane & 15;
    int koff  = (lane >> 4) << 3;
    const float* p = Arow + (size_t)row * C_ + kstep * 32 + koff + (hi << 4);
    unsigned short* q = sp + ((kstep * 32 + lane) << 4) + (hi << 3);
#pragma unroll
    for (int j = 0; j < 8; ++j) q[j] = f32_to_bf16_bits(p[j]);
}

// C/D 16x16 f32 store: lane col = lane&15, VGPR r -> row r (+8 for lanes 16-31).
__device__ __forceinline__ void store_c_frag(float* __restrict__ Cout, int ldc,
                                             int rowBase, int colBase, const v8f& c) {
    int lane = threadIdx.x & 31;
    int col  = colBase + (lane & 15);
    int roff = (lane >> 4) << 3;
#pragma unroll
    for (int r = 0; r < 8; ++r)
        Cout[(size_t)(rowBase + roff + r) * ldc + col] = c[r];
}

__global__ void k_zero(float* __restrict__ p, int n) {
    int i = blockIdx.x * blockDim.x + threadIdx.x;
    if (i < n) p[i] = 0.f;
}

// ---------- h = x @ W_emb + b_emb  ([N,32]@[32,128], K = one WMMA step) ----------
__global__ void k_embed(const float* __restrict__ x, const v16bf* __restrict__ Bp,
                        const float* __restrict__ bemb, float* __restrict__ h) {
    __shared__ v16bf aStage[32];                     // one kstep of A frags (1KB)
    int nodeBase = blockIdx.x * 16;
    if (threadIdx.x < 64) {                          // 32 lanes x 2 halves
        int u = threadIdx.x;
        int hi = u & 1, lane = u >> 1;
        int row = lane & 15, koff = (lane >> 4) << 3;
        const float* p = x + (size_t)(nodeBase + row) * F_IN_ + koff + (hi << 4);
        unsigned short* q = (unsigned short*)&aStage[lane] + (hi << 3);
#pragma unroll
        for (int j = 0; j < 8; ++j) q[j] = f32_to_bf16_bits(p[j]);
    }
    __syncthreads();
    int wave = threadIdx.x >> 5;
    int lane = threadIdx.x & 31;
    v16bf a = aStage[lane];
    v16bf b = Bp[wave * 32 + lane];                  // tile = wave, ksteps = 1
    v8f c = {};
    c = __builtin_amdgcn_wmma_f32_16x16x32_bf16(false, a, false, b, (short)0, c, false, false);
    float bias = bemb[wave * 16 + (lane & 15)];
#pragma unroll
    for (int r = 0; r < 8; ++r) c[r] += bias;
    store_c_frag(h, C_, nodeBase, wave * 16, c);
}

// ---------- m = h @ Wl  ([N,128]@[128,128]) ----------
__global__ void k_matmul(const float* __restrict__ A, const v16bf* __restrict__ Bp,
                         float* __restrict__ Cout) {
    __shared__ v16bf aStage[4 * 32];                 // 4 ksteps of A frags (4KB)
    int nodeBase = blockIdx.x * 16;
    stage_a_unit(A + (size_t)nodeBase * C_, (unsigned short*)aStage, threadIdx.x);
    __syncthreads();
    int wave = threadIdx.x >> 5;
    int lane = threadIdx.x & 31;
    v8f c = {};
#pragma unroll
    for (int k = 0; k < 4; ++k) {
        v16bf a = aStage[k * 32 + lane];
        v16bf b = Bp[(wave * 4 + k) * 32 + lane];
        if (k < 3) __builtin_prefetch(&Bp[(wave * 4 + k + 1) * 32 + lane], 0, 3);
        c = __builtin_amdgcn_wmma_f32_16x16x32_bf16(false, a, false, b, (short)0, c, false, false);
    }
    store_c_frag(Cout, C_, nodeBase, wave * 16, c);
}

// ---------- edge gather * w -> scatter-add (f32 atomics, L2-resident) ----------
__global__ void k_edge(const float4* __restrict__ m4, const int* __restrict__ src,
                       const int* __restrict__ dst, const float* __restrict__ ew,
                       float* __restrict__ agg) {
    int gid = blockIdx.x * blockDim.x + threadIdx.x;
    int e = gid >> 5;
    if (e >= EDGES) return;
    int lane = threadIdx.x & 31;
    int s = src[e], d = dst[e];
    float w = ew[e];
    float4 v = m4[(size_t)s * 32 + lane];
    float* p = agg + (size_t)d * C_ + lane * 4;
    atomicAdd(p + 0, v.x * w);
    atomicAdd(p + 1, v.y * w);
    atomicAdd(p + 2, v.z * w);
    atomicAdd(p + 3, v.w * w);
}

// ---------- fused GRU: gi = agg@W_ih^T+b_ih, gh = h@W_hh^T+b_hh, gates ----------
__global__ void k_gru(const float* __restrict__ agg, float* __restrict__ h,
                      const v16bf* __restrict__ BihP, const v16bf* __restrict__ BhhP,
                      const float* __restrict__ bih, const float* __restrict__ bhh) {
    __shared__ v16bf aStage[2][4 * 32];              // [0]=agg frags, [1]=h frags (8KB)
    __shared__ float gi_s[16][3 * C_ + 4];
    __shared__ float gh_s[16][3 * C_ + 4];
    int nodeBase = blockIdx.x * 16;

    // stage both A tiles (2 x 256 fill units over 256 threads)
    stage_a_unit(agg + (size_t)nodeBase * C_, (unsigned short*)&aStage[0][0], threadIdx.x);
    stage_a_unit(h   + (size_t)nodeBase * C_, (unsigned short*)&aStage[1][0], threadIdx.x);
    __syncthreads();

    int wave = threadIdx.x >> 5;
    int lane = threadIdx.x & 31;
#pragma unroll
    for (int t = 0; t < 6; ++t) {
        int tile = wave * 6 + t;                     // 0..47: 24 gi tiles, 24 gh tiles
        bool isGh = tile >= 24;
        int bt = isGh ? tile - 24 : tile;
        const v16bf* Bp = isGh ? BhhP : BihP;
        v8f c = {};
#pragma unroll
        for (int k = 0; k < 4; ++k) {
            v16bf a = aStage[isGh ? 1 : 0][k * 32 + lane];
            v16bf b = Bp[(bt * 4 + k) * 32 + lane];
            if (k < 3) __builtin_prefetch(&Bp[(bt * 4 + k + 1) * 32 + lane], 0, 3);
            c = __builtin_amdgcn_wmma_f32_16x16x32_bf16(false, a, false, b, (short)0, c, false, false);
        }
        int col  = bt * 16 + (lane & 15);
        int roff = (lane >> 4) << 3;
        float (*S)[3 * C_ + 4] = isGh ? gh_s : gi_s;
#pragma unroll
        for (int r = 0; r < 8; ++r)
            S[roff + r][col] = c[r];
    }
    __syncthreads();

    int node = threadIdx.x >> 4;                     // 0..15
    int cb   = (threadIdx.x & 15) * 8;               // 8 channels each
#pragma unroll
    for (int j = 0; j < 8; ++j) {
        int cc = cb + j;
        float i_r = gi_s[node][cc]        + bih[cc];
        float i_z = gi_s[node][cc +   C_] + bih[cc +   C_];
        float i_n = gi_s[node][cc + 2*C_] + bih[cc + 2*C_];
        float h_r = gh_s[node][cc]        + bhh[cc];
        float h_z = gh_s[node][cc +   C_] + bhh[cc +   C_];
        float h_n = gh_s[node][cc + 2*C_] + bhh[cc + 2*C_];
        float hp  = h[(size_t)(nodeBase + node) * C_ + cc];
        float r = 1.f / (1.f + __expf(-(i_r + h_r)));
        float z = 1.f / (1.f + __expf(-(i_z + h_z)));
        float n = tanhf(i_n + r * h_n);
        h[(size_t)(nodeBase + node) * C_ + cc] = (1.f - z) * n + z * hp;
    }
}

// ---------- readout: graph_prop[g] += h[i] . W_prop + b_prop ----------
__global__ void k_readout(const float* __restrict__ h, const float* __restrict__ Wprop,
                          const float* __restrict__ bprop, const int* __restrict__ batch,
                          float* __restrict__ out) {
    int gid = blockIdx.x * blockDim.x + threadIdx.x;
    int node = gid >> 5;
    if (node >= N_NODES) return;
    int lane = threadIdx.x & 31;
    const float4* h4 = (const float4*)(h + (size_t)node * C_);
    const float4* w4 = (const float4*)Wprop;
    float4 a = h4[lane], b = w4[lane];
    float v = a.x * b.x + a.y * b.y + a.z * b.z + a.w * b.w;
#pragma unroll
    for (int off = 16; off > 0; off >>= 1)
        v += __shfl_xor(v, off, 32);
    if (lane == 0) atomicAdd(&out[batch[node]], v + bprop[0]);
}

extern "C" void kernel_launch(void* const* d_in, const int* in_sizes, int n_in,
                              void* d_out, int out_size, void* d_ws, size_t ws_size,
                              hipStream_t stream) {
    const float* x      = (const float*)d_in[0];
    const int*   ei     = (const int*)  d_in[1];
    const float* ew     = (const float*)d_in[2];
    const int*   batch  = (const int*)  d_in[3];
    const float* W_emb  = (const float*)d_in[4];
    const float* b_emb  = (const float*)d_in[5];
    const float* W      = (const float*)d_in[6];
    const float* W_ih   = (const float*)d_in[7];
    const float* W_hh   = (const float*)d_in[8];
    const float* b_ih   = (const float*)d_in[9];
    const float* b_hh   = (const float*)d_in[10];
    const float* W_prop = (const float*)d_in[11];
    const float* b_prop = (const float*)d_in[12];
    float* out = (float*)d_out;

    // packed fragment counts (halves)
    const int PEMB_N = 8 * 1 * 32 * 16;              // 4096
    const int PW_N   = 8 * 4 * 32 * 16;              // 16384 per layer
    const int PIH_N  = 24 * 4 * 32 * 16;             // 49152

    char* ws = (char*)d_ws;
    float* h   = (float*)ws;  ws += (size_t)N_NODES * C_ * sizeof(float);
    float* m   = (float*)ws;  ws += (size_t)N_NODES * C_ * sizeof(float);
    float* agg = (float*)ws;  ws += (size_t)N_NODES * C_ * sizeof(float);
    unsigned short* Pemb = (unsigned short*)ws;  ws += (size_t)PEMB_N * 2;
    unsigned short* PW   = (unsigned short*)ws;  ws += (size_t)L_ * PW_N * 2;
    unsigned short* Pih  = (unsigned short*)ws;  ws += (size_t)PIH_N * 2;
    unsigned short* Phh  = (unsigned short*)ws;  ws += (size_t)PIH_N * 2;

    const int* src = ei;
    const int* dst = ei + EDGES;

    k_zero<<<(G_ + 255) / 256, 256, 0, stream>>>(out, G_);
    k_packB<<<(PEMB_N + 255) / 256, 256, 0, stream>>>(W_emb, Pemb, 8, 1, C_, 0, PEMB_N);
    for (int l = 0; l < L_; ++l)
        k_packB<<<(PW_N + 255) / 256, 256, 0, stream>>>(W + (size_t)l * C_ * C_,
                                                        PW + (size_t)l * PW_N, 8, 4, C_, 0, PW_N);
    k_packB<<<(PIH_N + 255) / 256, 256, 0, stream>>>(W_ih, Pih, 24, 4, C_, 1, PIH_N);
    k_packB<<<(PIH_N + 255) / 256, 256, 0, stream>>>(W_hh, Phh, 24, 4, C_, 1, PIH_N);

    const int nBlocks = N_NODES / 16;                // 6250 (exact)
    k_embed<<<nBlocks, 256, 0, stream>>>(x, (const v16bf*)Pemb, b_emb, h);

    for (int l = 0; l < L_; ++l) {
        k_matmul<<<nBlocks, 256, 0, stream>>>(h, (const v16bf*)(PW + (size_t)l * PW_N), m);
        k_zero<<<(N_NODES * C_ + 255) / 256, 256, 0, stream>>>(agg, N_NODES * C_);
        k_edge<<<(EDGES * 32) / 256, 256, 0, stream>>>((const float4*)m, src, dst, ew, agg);
        k_gru<<<nBlocks, 256, 0, stream>>>(agg, h, (const v16bf*)Pih, (const v16bf*)Phh,
                                           b_ih, b_hh);
    }

    k_readout<<<(N_NODES * 32 + 255) / 256, 256, 0, stream>>>(h, W_prop, b_prop, batch, out);
}
